// DihedralHandler_54623394070830
// MI455X (gfx1250) — matrix operand order repaired and checked by reading, not compiled, not run on hardware
//
#include <hip/hip_runtime.h>

#define LREL 0.01f
#define SLP  0.2f
#define PI_F 3.14159265358979323846f

typedef __attribute__((ext_vector_type(16))) __bf16 v16bf;
typedef __attribute__((ext_vector_type(8)))  float  v8f;

union U16bf { uint4 u[2]; v16bf v; };
union U8bf  { __bf16 h[8]; uint4 u; };

// ================= generic bf16 WMMA GEMM (contiguous A) =================
// C[m,n] = epi( sum_k A[m,k]*W[n,k] + bias[n] (+C if accum) ) * (gate?gate[m,n]:1)
// REQUIRES: gridDim.x*64 == M, gridDim.y*64 == N, K % 32 == 0, 16B-aligned rows.
__global__ __launch_bounds__(128) void gemm_wmma(
    const float* __restrict__ A, long ArS,
    const float* __restrict__ W, long WrS,
    float* __restrict__ C, long CrS,
    const float* __restrict__ bias,
    const float* __restrict__ gate, int gateS,
    int K, int accum, int epi)
{
  __shared__ __bf16 As[64][40];
  __shared__ __bf16 Ws[64][40];
  const int tid  = threadIdx.x;
  const int wave = tid >> 5, lane = tid & 31;
  const int hi = lane >> 4, lr = lane & 15;
  const long bm = (long)blockIdx.x * 64;
  const long bn = (long)blockIdx.y * 64;
  const int lrow = tid >> 1;         // 0..63 : tile row this thread loads
  const int lcol = (tid & 1) * 16;   // 0 / 16: 16 contiguous k per thread

  const float* Arow = A + (bm + lrow) * ArS + lcol;
  const float* Wrow = W + (bn + lrow) * WrS + lcol;

  v8f zero = {0.f,0.f,0.f,0.f,0.f,0.f,0.f,0.f};
  v8f acc[4]; acc[0]=zero; acc[1]=zero; acc[2]=zero; acc[3]=zero;

  for (int k0 = 0; k0 < K; k0 += 32) {
    {
      const float4* p = (const float4*)(Arow + k0);
      float4 f0 = p[0], f1 = p[1], f2 = p[2], f3 = p[3];
      U8bf lo, hq;
      lo.h[0]=(__bf16)f0.x; lo.h[1]=(__bf16)f0.y; lo.h[2]=(__bf16)f0.z; lo.h[3]=(__bf16)f0.w;
      lo.h[4]=(__bf16)f1.x; lo.h[5]=(__bf16)f1.y; lo.h[6]=(__bf16)f1.z; lo.h[7]=(__bf16)f1.w;
      hq.h[0]=(__bf16)f2.x; hq.h[1]=(__bf16)f2.y; hq.h[2]=(__bf16)f2.z; hq.h[3]=(__bf16)f2.w;
      hq.h[4]=(__bf16)f3.x; hq.h[5]=(__bf16)f3.y; hq.h[6]=(__bf16)f3.z; hq.h[7]=(__bf16)f3.w;
      uint4* d = (uint4*)&As[lrow][lcol];
      d[0] = lo.u; d[1] = hq.u;
    }
    {
      const float4* q = (const float4*)(Wrow + k0);
      float4 g0 = q[0], g1 = q[1], g2 = q[2], g3 = q[3];
      U8bf lo, hq;
      lo.h[0]=(__bf16)g0.x; lo.h[1]=(__bf16)g0.y; lo.h[2]=(__bf16)g0.z; lo.h[3]=(__bf16)g0.w;
      lo.h[4]=(__bf16)g1.x; lo.h[5]=(__bf16)g1.y; lo.h[6]=(__bf16)g1.z; lo.h[7]=(__bf16)g1.w;
      hq.h[0]=(__bf16)g2.x; hq.h[1]=(__bf16)g2.y; hq.h[2]=(__bf16)g2.z; hq.h[3]=(__bf16)g2.w;
      hq.h[4]=(__bf16)g3.x; hq.h[5]=(__bf16)g3.y; hq.h[6]=(__bf16)g3.z; hq.h[7]=(__bf16)g3.w;
      uint4* d = (uint4*)&Ws[lrow][lcol];
      d[0] = lo.u; d[1] = hq.u;
    }
    __syncthreads();
    U16bf af;
    {
      int mrow = wave*16 + lr;   // ISA A 16x32: lane hi half owns K {8..15,24..31}
      af.u[0] = *(const uint4*)&As[mrow][hi*8];
      af.u[1] = *(const uint4*)&As[mrow][16 + hi*8];
    }
    #pragma unroll
    for (int j = 0; j < 4; ++j) {
      U16bf bf;                  // ISA B 32x16: lane hi half owns K {8..15,24..31}
      bf.u[0] = *(const uint4*)&Ws[j*16 + lr][hi*8];
      bf.u[1] = *(const uint4*)&Ws[j*16 + lr][16 + hi*8];
      acc[j] = __builtin_amdgcn_wmma_f32_16x16x32_bf16(false, af.v, false, bf.v,
                                                       (short)0, acc[j], false, false);
    }
    __syncthreads();
  }
  #pragma unroll
  for (int j = 0; j < 4; ++j) {
    long n = bn + j*16 + lr;
    float bv = bias ? bias[n] : 0.f;
    #pragma unroll
    for (int r = 0; r < 8; ++r) {
      long m = bm + wave*16 + hi*8 + r;
      float v = acc[j][r] + bv;
      long ci = m*CrS + n;
      if (accum) v += C[ci];
      if (epi == 1) v = v > 0.f ? v : LREL*v;
      else if (epi == 2) v = 1.f/(1.f + __expf(-v));
      if (gate) v *= gate[m*(long)gateS + n];
      C[ci] = v;
    }
  }
}

// ===== fused 3-component vector-channel GEMM: C[m,n,d] = sum_k A[m,k,d]*W[n,k] =====
// A,C are (M,64,3) f32; W is (64,K) row-major; N fixed at 64; K % 32 == 0 (K=64 here).
// Optional gate[m,n] applied to all 3 components.
__global__ __launch_bounds__(128) void gemm_wmma_v3(
    const float* __restrict__ A,
    const float* __restrict__ W,
    float* __restrict__ C,
    const float* __restrict__ gate,
    int K)
{
  __shared__ __bf16 As[3][64][40];
  __shared__ __bf16 Ws[64][40];
  const int tid  = threadIdx.x;
  const int wave = tid >> 5, lane = tid & 31;
  const int hi = lane >> 4, lr = lane & 15;
  const long bm = (long)blockIdx.x * 64;
  const int lrow = tid >> 1;       // tile row this thread loads
  const int half = tid & 1;        // 16-k half

  const float* Arow = A + (bm + lrow)*192 + half*48;   // 16 k * 3 comps contiguous
  const float* Wrow = W + lrow*(long)K + half*16;

  v8f zero = {0.f,0.f,0.f,0.f,0.f,0.f,0.f,0.f};
  v8f acc[3][4];
  #pragma unroll
  for (int d = 0; d < 3; ++d)
    #pragma unroll
    for (int j = 0; j < 4; ++j) acc[d][j] = zero;

  for (int k0 = 0; k0 < K; k0 += 32) {
    // 48 contiguous floats: k-positions half*16..+15, interleaved xyz
    float fv[48];
    {
      const float4* p = (const float4*)(Arow + k0*3);
      #pragma unroll
      for (int q = 0; q < 12; ++q) {
        float4 f = p[q];
        fv[q*4+0]=f.x; fv[q*4+1]=f.y; fv[q*4+2]=f.z; fv[q*4+3]=f.w;
      }
    }
    #pragma unroll
    for (int d = 0; d < 3; ++d) {
      U8bf lo, hq;
      #pragma unroll
      for (int i = 0; i < 8; ++i) {
        lo.h[i] = (__bf16)fv[i*3 + d];
        hq.h[i] = (__bf16)fv[(8+i)*3 + d];
      }
      uint4* dst = (uint4*)&As[d][lrow][half*16];
      dst[0] = lo.u; dst[1] = hq.u;
    }
    {
      const float4* q = (const float4*)(Wrow + k0);
      float4 g0 = q[0], g1 = q[1], g2 = q[2], g3 = q[3];
      U8bf lo, hq;
      lo.h[0]=(__bf16)g0.x; lo.h[1]=(__bf16)g0.y; lo.h[2]=(__bf16)g0.z; lo.h[3]=(__bf16)g0.w;
      lo.h[4]=(__bf16)g1.x; lo.h[5]=(__bf16)g1.y; lo.h[6]=(__bf16)g1.z; lo.h[7]=(__bf16)g1.w;
      hq.h[0]=(__bf16)g2.x; hq.h[1]=(__bf16)g2.y; hq.h[2]=(__bf16)g2.z; hq.h[3]=(__bf16)g2.w;
      hq.h[4]=(__bf16)g3.x; hq.h[5]=(__bf16)g3.y; hq.h[6]=(__bf16)g3.z; hq.h[7]=(__bf16)g3.w;
      uint4* d = (uint4*)&Ws[lrow][half*16];
      d[0] = lo.u; d[1] = hq.u;
    }
    __syncthreads();
    U16bf af[3];
    {
      int mrow = wave*16 + lr;
      #pragma unroll
      for (int d = 0; d < 3; ++d) {
        af[d].u[0] = *(const uint4*)&As[d][mrow][hi*8];
        af[d].u[1] = *(const uint4*)&As[d][mrow][16 + hi*8];
      }
    }
    #pragma unroll
    for (int j = 0; j < 4; ++j) {
      U16bf bf;
      bf.u[0] = *(const uint4*)&Ws[j*16 + lr][hi*8];
      bf.u[1] = *(const uint4*)&Ws[j*16 + lr][16 + hi*8];
      #pragma unroll
      for (int d = 0; d < 3; ++d)
        acc[d][j] = __builtin_amdgcn_wmma_f32_16x16x32_bf16(false, af[d].v, false, bf.v,
                                                            (short)0, acc[d][j], false, false);
    }
    __syncthreads();
  }
  // epilogue: 3 adjacent floats per (m,n) -> coalesced b96 stores
  #pragma unroll
  for (int j = 0; j < 4; ++j) {
    int n = j*16 + lr;
    #pragma unroll
    for (int r = 0; r < 8; ++r) {
      long m = bm + wave*16 + hi*8 + r;
      float g = gate ? gate[m*64 + n] : 1.f;
      float* cp = C + m*192 + (long)n*3;
      cp[0] = acc[0][j][r]*g;
      cp[1] = acc[1][j][r]*g;
      cp[2] = acc[2][j][r]*g;
    }
  }
}

// ================= small kernels =================
__global__ void knorm(const float* __restrict__ vi, float* __restrict__ nrm, long cnt) {
  long i = (long)blockIdx.x*blockDim.x + threadIdx.x;
  if (i >= cnt) return;
  float a = vi[i*3], b = vi[i*3+1], c = vi[i*3+2];
  nrm[i] = sqrtf(a*a + b*b + c*c);
}

__global__ void kedge_geom(const int* __restrict__ ei, const float* __restrict__ pos,
                           const float* __restrict__ ef, float* __restrict__ dist,
                           float* __restrict__ ud, float* __restrict__ Cc,
                           float* __restrict__ unit, float* __restrict__ es, long E_) {
  long e = (long)blockIdx.x*blockDim.x + threadIdx.x;
  if (e >= E_) return;
  long i0 = ei[e], i1 = ei[E_ + e];
  float dx = pos[i0*3]-pos[i1*3], dy = pos[i0*3+1]-pos[i1*3+1], dz = pos[i0*3+2]-pos[i1*3+2];
  float d = sqrtf(dx*dx+dy*dy+dz*dz);
  dist[e] = d;
  float inv = 1.f/(d + 1e-7f);
  unit[e*3] = dx*inv; unit[e*3+1] = dy*inv; unit[e*3+2] = dz*inv;
  ud[e] = d*inv;
  Cc[e] = 0.5f*(__cosf(d*PI_F/10.f)+1.f) * (d <= 10.f ? 1.f : 0.f);
  const float delta = 10.f/26.f;
  const float coeff = -0.5f/(delta*delta);
  for (int j = 0; j < 27; ++j) { float t = d - (float)j*delta; es[e*32+j] = __expf(coeff*t*t); }
  for (int j = 0; j < 5;  ++j) es[e*32+27+j] = ef[e*5+j];
}

// rank-1 edge_gv precompute: pre[0:32]=u_s, pre[32:64]=w2v
__global__ void kedge_pre(const float* __restrict__ w_v1, const float* __restrict__ w_v2,
                          const float* __restrict__ w_s, const float* __restrict__ w_ve,
                          float* __restrict__ pre) {
  __shared__ float wv[32];
  int t = threadIdx.x;
  float s = 0.f;
  for (int c = 0; c < 32; ++c) s += w_v1[t*32+c]*w_ve[c];
  wv[t] = s; __syncthreads();
  float us = 0.f, w2 = 0.f;
  for (int o = 0; o < 32; ++o) us += w_s[t*64+o]*fabsf(wv[o]);
  for (int c = 0; c < 32; ++c) w2 += w_v2[t*32+c]*wv[c];
  pre[t] = us; pre[32+t] = w2;
}

// fused edge_gv + vn_leaky_relu (rank-1 in unit direction): one wave per edge
__global__ __launch_bounds__(256) void kedge_small(
    const float* __restrict__ es_in, const float* __restrict__ ud,
    const float* __restrict__ w_s, const float* __restrict__ w_g,
    const float* __restrict__ b_g, const float* __restrict__ dirW,
    const float* __restrict__ pre, float* __restrict__ es_act,
    float* __restrict__ coefb, long E_)
{
  int wv = threadIdx.x >> 5, lane = threadIdx.x & 31;
  long e = (long)blockIdx.x*8 + wv;
  bool ok = e < E_;
  __shared__ float sA[8][33], sB[8][33];
  sA[wv][lane] = ok ? es_in[e*32+lane] : 0.f;
  __syncthreads();
  float udv = ok ? ud[e] : 0.f;
  float acc = udv * pre[lane];
  for (int c = 0; c < 32; ++c) acc += sA[wv][c]*w_s[lane*64 + 32 + c];
  sB[wv][lane] = acc; __syncthreads();
  float g = b_g[lane];
  for (int c = 0; c < 32; ++c) g += sB[wv][c]*w_g[lane*32+c];
  g = 1.f/(1.f + __expf(-g));
  float a = g * pre[32+lane];
  sA[wv][lane] = a; __syncthreads();
  float bc = 0.f;
  for (int c = 0; c < 32; ++c) bc += dirW[lane*32+c]*sA[wv][c];
  if (ok) {
    float u2 = udv*udv;
    float dot = a*bc*u2, dsq = bc*bc*u2;
    float f = (dot >= 0.f) ? 0.f : dot/(dsq + 1e-6f);
    coefb[e*32+lane] = a - (1.f-SLP)*f*bc;
    es_act[e*32+lane] = acc > 0.f ? acc : LREL*acc;
  }
}

// y_s / y_v assembly (gather node features)
__global__ __launch_bounds__(256) void ky(const int* __restrict__ col,
    const float* __restrict__ ns, const float* __restrict__ nv,
    const float* __restrict__ sg, const float* __restrict__ e2n,
    const float* __restrict__ nn2e, const float* __restrict__ wc,
    const float* __restrict__ unit, float* __restrict__ y_s, float* __restrict__ y_v) {
  long e = blockIdx.x; int t = threadIdx.x;
  long c = col[e];
  y_s[e*256+t] = ns[c*256+t]*sg[e*256+t];
  if (t < 192) {
    int h = t/3, d = t - h*3;
    y_v[e*192+t] = e2n[e*64+h]*nv[c*192+t] + nn2e[c*64+h]*wc[e*64+h]*unit[e*3+d];
  }
}

__global__ __launch_bounds__(256) void kscatter(const int* __restrict__ row,
    const float* __restrict__ os_e, const float* __restrict__ ov_e,
    const float* __restrict__ Cc, float* __restrict__ aggr_s, float* __restrict__ aggr_v) {
  long e = blockIdx.x; int t = threadIdx.x;
  long r = row[e]; float c = Cc[e];
  atomicAdd(&aggr_s[r*256+t], os_e[e*256+t]*c);
  if (t < 192) atomicAdd(&aggr_v[r*192+t], ov_e[e*192+t]*c);
}

// dual LayerNorm (scalar 256 / vector 64x3) + leaky on scalars
__global__ __launch_bounds__(256) void kln(const float* __restrict__ cs, const float* __restrict__ as_,
    const float* __restrict__ cv, const float* __restrict__ av,
    const float* __restrict__ gs, const float* __restrict__ bs,
    const float* __restrict__ gv, const float* __restrict__ bv,
    float* __restrict__ s_act, float* __restrict__ ov) {
  long n = blockIdx.x; int t = threadIdx.x;
  __shared__ float red[256]; __shared__ float st[4];
  float xs = cs[n*256+t] + as_[n*256+t];
  red[t] = xs; __syncthreads();
  for (int o = 128; o > 0; o >>= 1) { if (t < o) red[t] += red[t+o]; __syncthreads(); }
  if (t == 0) st[0] = red[0]*(1.f/256.f);
  __syncthreads();
  red[t] = xs*xs; __syncthreads();
  for (int o = 128; o > 0; o >>= 1) { if (t < o) red[t] += red[t+o]; __syncthreads(); }
  if (t == 0) st[1] = red[0]*(1.f/256.f) - st[0]*st[0];
  __syncthreads();
  float v = (xs - st[0])*rsqrtf(st[1] + 1e-5f)*gs[t] + bs[t];
  s_act[n*256+t] = v > 0.f ? v : LREL*v;
  float xv = (t < 192) ? cv[n*192+t] + av[n*192+t] : 0.f;
  red[t] = xv; __syncthreads();
  for (int o = 128; o > 0; o >>= 1) { if (t < o) red[t] += red[t+o]; __syncthreads(); }
  if (t == 0) st[2] = red[0]*(1.f/192.f);
  __syncthreads();
  red[t] = xv*xv; __syncthreads();
  for (int o = 128; o > 0; o >>= 1) { if (t < o) red[t] += red[t+o]; __syncthreads(); }
  if (t == 0) st[3] = red[0]*(1.f/192.f) - st[2]*st[2];
  __syncthreads();
  if (t < 192) ov[n*192+t] = (xv - st[2])*rsqrtf(st[3] + 1e-5f)*gv[t] + bv[t];
}

__global__ void kvact(const float* __restrict__ x, const float* __restrict__ dd,
                      float* __restrict__ out, long cnt) {
  long i = (long)blockIdx.x*blockDim.x + threadIdx.x;
  if (i >= cnt) return;
  float x0=x[i*3],x1=x[i*3+1],x2=x[i*3+2];
  float d0=dd[i*3],d1=dd[i*3+1],d2=dd[i*3+2];
  float dot = x0*d0+x1*d1+x2*d2;
  float dsq = d0*d0+d1*d1+d2*d2;
  float f = (dot >= 0.f) ? 0.f : (1.f-SLP)*dot/(dsq + 1e-6f);
  out[i*3]   = x0 - f*d0;
  out[i*3+1] = x1 - f*d1;
  out[i*3+2] = x2 - f*d2;
}

__global__ void kadd(float* __restrict__ dst, const float* __restrict__ src, long n) {
  long i = (long)blockIdx.x*blockDim.x + threadIdx.x;
  if (i < n) dst[i] += src[i];
}

__global__ __launch_bounds__(256) void kmol(const float* __restrict__ h_s,
    const int* __restrict__ lig, const int* __restrict__ bm, float* __restrict__ mol) {
  long l = blockIdx.x; int t = threadIdx.x;
  atomicAdd(&mol[(long)bm[l]*256 + t], h_s[(long)lig[l]*256 + t]);
}

__global__ __launch_bounds__(256) void kfeat(const float* __restrict__ h_s,
    const float* __restrict__ mol, const int* __restrict__ a, const int* __restrict__ b,
    float* __restrict__ feat) {
  long m = blockIdx.x; int t = threadIdx.x;
  feat[m*768 + t]       = h_s[(long)a[m]*256 + t];
  feat[m*768 + 256 + t] = h_s[(long)b[m]*256 + t];
  feat[m*768 + 512 + t] = mol[m*256 + t];
}

__global__ __launch_bounds__(256) void kalpha(const float* __restrict__ hid,
    const float* __restrict__ w2, const float* __restrict__ b2, float* __restrict__ out) {
  long m = blockIdx.x; int t = threadIdx.x;
  __shared__ float red[256];
  float s = 0.f;
  for (int k = t; k < 768; k += 256) s += hid[m*768+k]*w2[k];
  red[t] = s; __syncthreads();
  for (int o = 128; o > 0; o >>= 1) { if (t < o) red[t] += red[t+o]; __syncthreads(); }
  if (t == 0) out[m] = red[0] + b2[0];
}

__global__ void krot(const float* __restrict__ pos, const int* __restrict__ a,
    const int* __restrict__ b, const int* __restrict__ bnx, const int* __restrict__ bbn,
    const float* __restrict__ alpha, float* __restrict__ out, long K_) {
  long k = (long)blockIdx.x*blockDim.x + threadIdx.x;
  if (k >= K_) return;
  long m = bbn[k];
  long ai = a[m], bi = b[m], ni = bnx[k];
  float ax = pos[ai*3]-pos[bi*3], ay = pos[ai*3+1]-pos[bi*3+1], az = pos[ai*3+2]-pos[bi*3+2];
  float nr = sqrtf(ax*ax+ay*ay+az*az) + 1e-9f;
  float x = ax/nr, y = ay/nr, z = az/nr;
  float al = alpha[m];
  float c = __cosf(al), s = __sinf(al), tt = 1.f - c;
  float px = pos[ni*3]-pos[ai*3], py = pos[ni*3+1]-pos[ai*3+1], pz = pos[ni*3+2]-pos[ai*3+2];
  float rx = (tt*x*x+c)*px + (tt*x*y - s*z)*py + (tt*x*z + s*y)*pz;
  float ry = (tt*x*y+s*z)*px + (tt*y*y+c)*py + (tt*y*z - s*x)*pz;
  float rz = (tt*x*z-s*y)*px + (tt*y*z + s*x)*py + (tt*z*z+c)*pz;
  out[k*3]   = rx + pos[ai*3];
  out[k*3+1] = ry + pos[ai*3+1];
  out[k*3+2] = rz + pos[ai*3+2];
}

// ================= host-side orchestration =================
struct GVo { long b_g, w_g, w_s, w_v1, w_v2; };
struct Cur { long c; long take(long n) { long o = c; c += n; return o; } };
static GVo takeGV(Cur& cu, long in_s, long in_v, long out_s, long out_v, long dh) {
  GVo g;
  g.b_g  = cu.take(out_v);
  g.w_g  = cu.take(out_v*out_s);
  g.w_s  = cu.take(out_s*(in_s+dh));
  g.w_v1 = cu.take(dh*in_v);
  g.w_v2 = cu.take(out_v*dh);
  return g;
}
struct BlkOff {
  long act_dir; GVo centroid;
  long ln_s_b, ln_s_g, ln_v_b, ln_v_g;
  long b_e2n, b_n2e, b_sca;
  GVo edge_gv; long edge_gv_dir;
  GVo node_gv; GVo out_gv;
  long w_e2n, w_edge_vn, w_n2e, w_sca;
  GVo out; long w_ve;
};
static BlkOff parse_block(Cur& cu) {
  BlkOff b;
  b.act_dir  = cu.take(64*64);
  b.centroid = takeGV(cu,256,64,256,64,64);
  b.ln_s_b = cu.take(256); b.ln_s_g = cu.take(256);
  b.ln_v_b = cu.take(192); b.ln_v_g = cu.take(192);
  b.b_e2n = cu.take(64); b.b_n2e = cu.take(64); b.b_sca = cu.take(256);
  b.edge_gv = takeGV(cu,32,32,32,32,32);
  b.edge_gv_dir = cu.take(1024);
  b.node_gv = takeGV(cu,256,64,256,64,64);
  b.out_gv  = takeGV(cu,256,64,256,64,64);
  b.w_e2n = cu.take(64*32); b.w_edge_vn = cu.take(64*32);
  b.w_n2e = cu.take(64*256); b.w_sca = cu.take(256*32);
  b.out = takeGV(cu,256,64,256,64,64);
  b.w_ve = cu.take(32);
  return b;
}

static inline void launch_gemm(hipStream_t st, const float* A, long ArS,
    const float* W, long WrS, float* C, long CrS,
    const float* bias, const float* gate, int gateS,
    long M, int Nn, int K, int accum, int epi) {
  dim3 g((unsigned)((M+63)/64), (unsigned)((Nn+63)/64)), blk(128);
  gemm_wmma<<<g, blk, 0, st>>>(A,ArS,W,WrS,C,CrS,bias,gate,gateS,K,accum,epi);
}

// full gv_linear: out_s, out_v(gated); vi/nrm/gate scratch sized for R rows
static void run_gv(hipStream_t st, const float* P, const GVo& g,
    const float* in_s, const float* in_v, float* out_s, float* out_v,
    float* vi, float* nrm, float* gate, long R) {
  gemm_wmma_v3<<<dim3((unsigned)(R/64)), dim3(128), 0, st>>>(in_v, P+g.w_v1, vi, nullptr, 64);
  long c64 = R*64;
  knorm<<<dim3((unsigned)((c64+255)/256)), dim3(256), 0, st>>>(vi, nrm, c64);
  launch_gemm(st, nrm,64,   P+g.w_s,   320, out_s,256, nullptr,nullptr,0, R,256,64,  0,0);
  launch_gemm(st, in_s,256, P+g.w_s+64,320, out_s,256, nullptr,nullptr,0, R,256,256, 1,0);
  launch_gemm(st, out_s,256,P+g.w_g,   256, gate,64,   P+g.b_g,nullptr,0, R,64,256,  0,2);
  gemm_wmma_v3<<<dim3((unsigned)(R/64)), dim3(128), 0, st>>>(vi, P+g.w_v2, out_v, gate, 64);
}

extern "C" void kernel_launch(void* const* d_in, const int* in_sizes, int n_in,
                              void* d_out, int out_size, void* d_ws, size_t ws_size,
                              hipStream_t stream) {
  const float* P    = (const float*)d_in[0];
  const float* sca  = (const float*)d_in[1];
  const float* vec  = (const float*)d_in[2];
  const float* pos  = (const float*)d_in[3];
  const int*   ei   = (const int*)d_in[4];
  const float* ef   = (const float*)d_in[5];
  const int*   ia   = (const int*)d_in[6];
  const int*   ibb  = (const int*)d_in[7];
  const int*   lig  = (const int*)d_in[8];
  const int*   bmol = (const int*)d_in[9];
  const int*   bnx  = (const int*)d_in[10];
  const int*   bbn  = (const int*)d_in[11];
  float* out = (float*)d_out;

  const long N = in_sizes[1]/256;
  const long E = in_sizes[4]/2;
  const long M = in_sizes[6];
  const long L = in_sizes[8];
  const long K = in_sizes[10];

  // param offsets (JAX tree: sorted dict keys; 'blocks' < 'mlp')
  Cur cu; cu.c = 0;
  BlkOff B[2]; B[0] = parse_block(cu); B[1] = parse_block(cu);
  long mb1 = cu.take(768), mb2 = cu.take(1), mw1 = cu.take(768*768), mw2 = cu.take(768);

  // workspace bump allocator (256B aligned blocks)
  char* wsc = (char*)d_ws;
  auto alloc = [&](long nf) -> float* {
    float* p = (float*)wsc; wsc += ((nf*4 + 255)/256)*256; return p;
  };
  float* h_s   = alloc(N*256); float* h_v   = alloc(N*192);
  float* dist  = alloc(E);     float* udb   = alloc(E);     float* Cc = alloc(E);
  float* unit  = alloc(3*E);   float* es0   = alloc(32*E);
  float* vi_n  = alloc(N*192); float* nrm_n = alloc(N*64);  float* gate_n = alloc(N*64);
  float* ns    = alloc(N*256); float* nv    = alloc(N*192);
  float* nn2e  = alloc(N*64);
  float* aggr_s= alloc(N*256); float* aggr_v= alloc(N*192);
  float* s_act = alloc(N*256); float* o_v   = alloc(N*192);
  float* dvb   = alloc(N*192); float* v_act = alloc(N*192);
  float* d_s   = alloc(N*256); float* d_v   = alloc(N*192);
  float* es_act= alloc(32*E);  float* coefb = alloc(32*E);
  float* sg    = alloc(256*E); float* e2n   = alloc(64*E);  float* wcb = alloc(64*E);
  float* y_s   = alloc(256*E); float* y_v   = alloc(192*E);
  float* vi_e  = alloc(192*E); float* nrm_e = alloc(64*E);  float* gate_e = alloc(64*E);
  float* pre   = alloc(64);
  float* mol   = alloc(M*256); float* feat  = alloc(M*768); float* hid = alloc(M*768);
  float* w1a   = alloc(768*768);   // aligned copy of mlp.w1 (param offset is 4B-aligned only)
  float* os_e  = sg;    // dead after ky
  float* ov_e  = y_s;   // dead after out_gv scalar accum GEMM

  hipMemcpyAsync(h_s, sca, sizeof(float)*N*256, hipMemcpyDeviceToDevice, stream);
  hipMemcpyAsync(h_v, vec, sizeof(float)*N*192, hipMemcpyDeviceToDevice, stream);
  hipMemcpyAsync(w1a, P+mw1, sizeof(float)*768*768, hipMemcpyDeviceToDevice, stream);

  kedge_geom<<<dim3((unsigned)((E+255)/256)), dim3(256), 0, stream>>>(
      ei, pos, ef, dist, udb, Cc, unit, es0, E);

  for (int bi = 0; bi < 2; ++bi) {
    const BlkOff& b = B[bi];
    // node_gv
    run_gv(stream, P, b.node_gv, h_s, h_v, ns, nv, vi_n, nrm_n, gate_n, N);
    // rank-1 edge path
    kedge_pre<<<1, 32, 0, stream>>>(P+b.edge_gv.w_v1, P+b.edge_gv.w_v2,
                                    P+b.edge_gv.w_s, P+b.w_ve, pre);
    kedge_small<<<dim3((unsigned)((E+7)/8)), dim3(256), 0, stream>>>(
        es0, udb, P+b.edge_gv.w_s, P+b.edge_gv.w_g, P+b.edge_gv.b_g,
        P+b.edge_gv_dir, pre, es_act, coefb, E);
    // edge/node projections
    launch_gemm(stream, es_act,32, P+b.w_sca,32,    sg,256,  P+b.b_sca,nullptr,0, E,256,32, 0,0);
    launch_gemm(stream, es_act,32, P+b.w_e2n,32,    e2n,64,  P+b.b_e2n,nullptr,0, E,64,32,  0,0);
    launch_gemm(stream, coefb,32,  P+b.w_edge_vn,32,wcb,64,  nullptr,nullptr,0,   E,64,32,  0,0);
    launch_gemm(stream, ns,256,    P+b.w_n2e,256,   nn2e,64, P+b.b_n2e,nullptr,0, N,64,256, 0,0);
    // assemble y
    ky<<<dim3((unsigned)E), dim3(256), 0, stream>>>(ei+E, ns, nv, sg, e2n, nn2e, wcb, unit, y_s, y_v);
    // out_gv over edges (big)
    run_gv(stream, P, b.out_gv, y_s, y_v, os_e, ov_e, vi_e, nrm_e, gate_e, E);
    // segment sum
    hipMemsetAsync(aggr_s, 0, sizeof(float)*N*256, stream);
    hipMemsetAsync(aggr_v, 0, sizeof(float)*N*192, stream);
    kscatter<<<dim3((unsigned)E), dim3(256), 0, stream>>>(ei, os_e, ov_e, Cc, aggr_s, aggr_v);
    // centroid (alias ns/nv as outputs; dead by now)
    run_gv(stream, P, b.centroid, h_s, h_v, ns, nv, vi_n, nrm_n, gate_n, N);
    // layernorm + leaky
    kln<<<dim3((unsigned)N), dim3(256), 0, stream>>>(ns, aggr_s, nv, aggr_v,
        P+b.ln_s_g, P+b.ln_s_b, P+b.ln_v_g, P+b.ln_v_b, s_act, o_v);
    // vn_leaky_relu(act_dir)
    gemm_wmma_v3<<<dim3((unsigned)(N/64)), dim3(128), 0, stream>>>(o_v, P+b.act_dir, dvb, nullptr, 64);
    kvact<<<dim3((unsigned)((N*64+255)/256)), dim3(256), 0, stream>>>(o_v, dvb, v_act, N*64);
    // out gv + residual
    run_gv(stream, P, b.out, s_act, v_act, d_s, d_v, vi_n, nrm_n, gate_n, N);
    kadd<<<dim3((unsigned)((N*256+255)/256)), dim3(256), 0, stream>>>(h_s, d_s, N*256);
    kadd<<<dim3((unsigned)((N*192+255)/256)), dim3(256), 0, stream>>>(h_v, d_v, N*192);
  }

  // head
  hipMemsetAsync(mol, 0, sizeof(float)*M*256, stream);
  kmol<<<dim3((unsigned)L), dim3(256), 0, stream>>>(h_s, lig, bmol, mol);
  kfeat<<<dim3((unsigned)M), dim3(256), 0, stream>>>(h_s, mol, ia, ibb, feat);
  launch_gemm(stream, feat,768, w1a,768, hid,768, P+mb1, nullptr,0, M,768,768, 0,1);
  kalpha<<<dim3((unsigned)M), dim3(256), 0, stream>>>(hid, P+mw2, P+mb2, out);
  krot<<<dim3((unsigned)((K+255)/256)), dim3(256), 0, stream>>>(
      pos, ia, ibb, bnx, bbn, out, out + M, K);

  (void)n_in; (void)out_size; (void)ws_size;
}
